// MaskedSymbolTransformer_44641890075303
// MI455X (gfx1250) — matrix operand order repaired and checked by reading, not compiled, not run on hardware
//
#include <hip/hip_runtime.h>
#include <math.h>

// ---------------- model constants ----------------
#define BB 2
#define CC 2
#define TT 4096
#define DD 512
#define HH 8
#define DH 64
#define NHASH 4
#define NB 64
#define NCH (NHASH * NB)      // 256 chunks of 64 in sorted domain
#define NHT (NHASH * TT)      // 16384 items per bh row
#define BHD (BB * HH)         // 16
#define DFF 2048
#define SS 64
#define DEPTH 6
#define EPSV 1e-5f
#define SELF_MASK -5e4f
#define MROWS (BB * TT)       // 8192 GEMM rows

// ---------------- CDNA5 vector types ----------------
typedef __bf16  bf16x16 __attribute__((ext_vector_type(16)));
typedef float   f32x8   __attribute__((ext_vector_type(8)));
typedef unsigned int u32x4 __attribute__((ext_vector_type(4)));

__device__ __forceinline__ unsigned short f2bfu(float f) {
    unsigned u = __float_as_uint(f);
    u += 0x7FFFu + ((u >> 16) & 1u);          // round-to-nearest-even
    return (unsigned short)(u >> 16);
}

__device__ __forceinline__ float gelu_exact(float x) {
    return 0.5f * x * (1.0f + erff(x * 0.70710678118654752f));
}

// Async copy 16B from global to LDS (gfx1250 GLOBAL_LOAD_ASYNC_TO_LDS_B128,
// tracked by ASYNCcnt). ldsaddr = low 32 bits of generic pointer (LDS aperture
// maps addr[31:0] to the LDS byte offset).
__device__ __forceinline__ void async_copy_b128(unsigned ldsaddr, const void* gaddr) {
    asm volatile("global_load_async_to_lds_b128 %0, %1, off"
                 :: "v"(ldsaddr), "v"(gaddr) : "memory");
}
__device__ __forceinline__ void async_wait0() {
    asm volatile("s_wait_asynccnt 0x0" ::: "memory");
}
__device__ __forceinline__ void async_wait4() {      // allow 4 newer ops in flight
    asm volatile("s_wait_asynccnt 0x4" ::: "memory");
}

// =====================================================================
// 0) Weight convert + transpose to bf16: Wt[n][k] = bf16(W[k][n])
// =====================================================================
__global__ void k_wT_bf16(const float* __restrict__ W, unsigned short* __restrict__ Wt,
                          int K, int N) {
    int idx = blockIdx.x * 256 + threadIdx.x;     // over K*N, coalesced read
    if (idx >= K * N) return;
    int n = idx % N, k = idx / N;
    Wt[(size_t)n * K + k] = f2bfu(W[idx]);
}

// =====================================================================
// 1) Embedding: 1x1 conv channel mix + sinusoidal PE  -> x1 and x2
// =====================================================================
__global__ void k_embed(const float* __restrict__ x, const float* __restrict__ cw,
                        const float* __restrict__ cb, float* __restrict__ x1,
                        float* __restrict__ x2) {
    int idx = blockIdx.x * 256 + threadIdx.x;       // over B*T*D
    if (idx >= BB * TT * DD) return;
    int d = idx % DD;
    int t = (idx / DD) % TT;
    int b = idx / (DD * TT);
    float h = x[(b * CC + 0) * TT + t] * cw[d * CC + 0]
            + x[(b * CC + 1) * TT + t] * cw[d * CC + 1] + cb[d];
    float di  = (float)(d & ~1);
    float div = __expf(di * (-9.210340371976184f / (float)DD));   // ln(10000)
    float ang = (float)t * div;
    h += (d & 1) ? __cosf(ang) : __sinf(ang);
    x1[idx] = h;
    x2[idx] = h;
}

// =====================================================================
// 2) LayerNorm: one wave per row of 512; emits bf16 (GEMM A operand)
// =====================================================================
__global__ void k_layernorm(const float* __restrict__ in, const float* __restrict__ g,
                            const float* __restrict__ be, unsigned short* __restrict__ out) {
    int row  = blockIdx.x * 8 + (threadIdx.x >> 5);
    int lane = threadIdx.x & 31;
    if (row >= MROWS) return;
    const float* r = in + (size_t)row * DD;
    float s = 0.f, s2 = 0.f;
    for (int i = lane; i < DD; i += 32) { float v = r[i]; s += v; s2 += v * v; }
    for (int m = 16; m >= 1; m >>= 1) {
        s  += __shfl_xor(s,  m, 32);
        s2 += __shfl_xor(s2, m, 32);
    }
    float mean = s * (1.0f / DD);
    float var  = s2 * (1.0f / DD) - mean * mean;
    float inv  = rsqrtf(var + EPSV);
    unsigned short* o = out + (size_t)row * DD;
    for (int i = lane; i < DD; i += 32)
        o[i] = f2bfu((r[i] - mean) * inv * g[i] + be[i]);
}

// =====================================================================
// 3) bf16 WMMA GEMM, double-buffered async global->LDS staging.
//    A16: [M][K] bf16 row-major.  Bt16: [N][K] bf16 (pre-transposed weight).
//    EPI: 0=none 1=+bias 2=+bias,gelu 3=+bias,+res   OUT16: bf16 vs f32 out
//    128 threads = 4 waves; tile 64x64, BK=32.
//    Each stage = 4 per-wave async ops (2 for A, 2 for B); the partial wait
//    s_wait_asynccnt 4 retires the older tile while the newer stays in flight.
// =====================================================================
#define GBM 64
#define GBN 64
#define GBK 32

template <int EPI, int OUT16>
__global__ void __launch_bounds__(128)
k_gemm_async(const unsigned short* __restrict__ A16,
             const unsigned short* __restrict__ Bt16,
             const float* __restrict__ bias, const float* __restrict__ res,
             void* __restrict__ Cout, int M, int N, int K) {
    __shared__ __align__(16) unsigned short As[2][GBM * GBK];  // [m][k] bf16, rows 64B
    __shared__ __align__(16) unsigned short Bs[2][GBN * GBK];  // [n][k] bf16, rows 64B

    int tid  = threadIdx.x;
    int wave = tid >> 5, lane = tid & 31;
    int group = lane >> 4, l16 = lane & 15;
    int m0 = blockIdx.y * GBM, n0 = blockIdx.x * GBN;

    // per-thread 16B chunk assignment: 256 chunks per 4KB tile, 2 per thread
    int c0  = tid;            // chunk ids c0, c0+128
    int r0  = c0 >> 2,        cc0 = c0 & 3;
    int r1  = (c0 + 128) >> 2, cc1 = (c0 + 128) & 3;

    unsigned ldsA0 = (unsigned)(size_t)(&As[0][0]);
    unsigned ldsB0 = (unsigned)(size_t)(&Bs[0][0]);

    auto stage = [&](int buf, int k0) {
        unsigned la = ldsA0 + buf * (GBM * GBK * 2);
        unsigned lb = ldsB0 + buf * (GBN * GBK * 2);
        async_copy_b128(la + c0 * 16,
                        A16 + (size_t)(m0 + r0) * K + k0 + cc0 * 8);
        async_copy_b128(la + (c0 + 128) * 16,
                        A16 + (size_t)(m0 + r1) * K + k0 + cc1 * 8);
        async_copy_b128(lb + c0 * 16,
                        Bt16 + (size_t)(n0 + r0) * K + k0 + cc0 * 8);
        async_copy_b128(lb + (c0 + 128) * 16,
                        Bt16 + (size_t)(n0 + r1) * K + k0 + cc1 * 8);
    };

    f32x8 acc[4];
    for (int a = 0; a < 4; ++a)
        for (int i = 0; i < 8; ++i) acc[a][i] = 0.f;

    int mrow = wave * 16 + l16;
    int nsteps = K / GBK;

    stage(0, 0);                                   // prime the pipe (ASYNCcnt=4)
    for (int ks = 0; ks < nsteps; ++ks) {
        int buf = ks & 1;
        bool has_next = (ks + 1 < nsteps);
        if (has_next) stage(buf ^ 1, (ks + 1) * GBK);   // overlap DMA with WMMA
        if (has_next) async_wait4();                    // older 4 ops retired
        else          async_wait0();
        __syncthreads();                                // publish DMA-written LDS

        const unsigned short* Ab = &As[buf][0];
        const unsigned short* Bb = &Bs[buf][0];
        // A fragment: 16x32 bf16, lane layout per CDNA5 ISA 7.12.2
        union { bf16x16 v; u32x4 q[2]; } af;
        af.q[0] = *(const u32x4*)&Ab[mrow * GBK + group * 8];
        af.q[1] = *(const u32x4*)&Ab[mrow * GBK + 16 + group * 8];

        #pragma unroll
        for (int a = 0; a < 4; ++a) {
            int n = a * 16 + l16;
            union { bf16x16 v; u32x4 q[2]; } bfv;
            bfv.q[0] = *(const u32x4*)&Bb[n * GBK + group * 8];
            bfv.q[1] = *(const u32x4*)&Bb[n * GBK + 16 + group * 8];
            acc[a] = __builtin_amdgcn_wmma_f32_16x16x32_bf16(
                false, af.v, false, bfv.v, (short)0, acc[a], false, false);
        }
        __syncthreads();     // all waves done reading buf before it is re-staged
    }

    // epilogue: C layout -> lane(group,l16) holds rows group*8+r, col l16 (per tile)
    #pragma unroll
    for (int a = 0; a < 4; ++a) {
        int n = n0 + a * 16 + l16;
        float bval = 0.f;
        if constexpr (EPI != 0) bval = bias[n];
        #pragma unroll
        for (int r = 0; r < 8; ++r) {
            int m = m0 + wave * 16 + group * 8 + r;
            float v = acc[a][r] + bval;
            if constexpr (EPI == 2) v = gelu_exact(v);
            if constexpr (EPI == 3) v += res[(size_t)m * N + n];
            if constexpr (OUT16) ((unsigned short*)Cout)[(size_t)m * N + n] = f2bfu(v);
            else                 ((float*)Cout)[(size_t)m * N + n] = v;
        }
    }
}

// =====================================================================
// 4) LSH hashing: bucket = argmax over [rot, -rot] per hash round
// =====================================================================
__global__ void k_hash(const float* __restrict__ qk, const float* __restrict__ rot,
                       int* __restrict__ buckets) {
    int idx = blockIdx.x * 256 + threadIdx.x;     // bh*NHASH*T
    if (idx >= BHD * NHASH * TT) return;
    int t  = idx % TT;
    int nh = (idx / TT) % NHASH;
    int bh = idx / (TT * NHASH);
    int b = bh / HH, hd = bh % HH;
    const float* q = qk + ((size_t)(b * TT + t)) * DD + hd * DH;
    float qv[DH];
    #pragma unroll
    for (int f = 0; f < DH; ++f) qv[f] = q[f];
    float rs[32];
    for (int i = 0; i < 32; ++i) {
        float s = 0.f;
        #pragma unroll
        for (int f = 0; f < DH; ++f) s += qv[f] * rot[(f * NHASH + nh) * 32 + i];
        rs[i] = s;
    }
    float best = rs[0]; int bi = 0;
    for (int i = 1; i < 32; ++i) if (rs[i]  > best) { best = rs[i];  bi = i; }
    for (int i = 0; i < 32; ++i) if (-rs[i] > best) { best = -rs[i]; bi = 32 + i; }
    buckets[idx] = bi + nh * NB;
}

// =====================================================================
// 5) Stable counting sort of (bucket, pos) per bh row: 256 bins
// =====================================================================
__global__ void k_hist(const int* __restrict__ buckets, int* __restrict__ Hc) {
    __shared__ int hist[256];
    int bh = blockIdx.y, tile = blockIdx.x, tid = threadIdx.x;
    hist[tid] = 0;
    __syncthreads();
    int bk = buckets[bh * NHT + tile * 256 + tid];
    atomicAdd(&hist[bk], 1);
    __syncthreads();
    Hc[(bh * 64 + tile) * 256 + tid] = hist[tid];
}

__global__ void k_binoff(const int* __restrict__ Hc, int* __restrict__ TOFF) {
    __shared__ int sc[256];
    int bh = blockIdx.x, bin = threadIdx.x;
    int run = 0;
    for (int c = 0; c < 64; ++c) {
        TOFF[(bh * 64 + c) * 256 + bin] = run;
        run += Hc[(bh * 64 + c) * 256 + bin];
    }
    sc[bin] = run;
    __syncthreads();
    for (int off = 1; off < 256; off <<= 1) {       // inclusive scan
        int v = (bin >= off) ? sc[bin - off] : 0;
        __syncthreads();
        sc[bin] += v;
        __syncthreads();
    }
    int start = sc[bin] - run;                      // exclusive
    for (int c = 0; c < 64; ++c)
        TOFF[(bh * 64 + c) * 256 + bin] += start;
}

__global__ void k_scatter(const int* __restrict__ buckets, const int* __restrict__ TOFF,
                          int* __restrict__ stv, int* __restrict__ undo) {
    __shared__ int lb[256];
    int bh = blockIdx.y, tile = blockIdx.x, tid = threadIdx.x;
    int i  = tile * 256 + tid;
    int bk = buckets[bh * NHT + i];
    lb[tid] = bk;
    __syncthreads();
    int rank = 0;
    for (int j = 0; j < tid; ++j) rank += (lb[j] == bk);
    int dst = TOFF[(bh * 64 + tile) * 256 + bk] + rank;
    stv[bh * NHT + dst] = i % TT;
    undo[bh * NHT + i]  = dst;
}

// =====================================================================
// 6) Fused LSH chunk attention: one block (128 thr, 4 waves) per (bh, chunk)
//    S = (Q*DH^-0.5) Khat^T  (64x128), mask self, softmax/LSE, O = P V (64x64)
// =====================================================================
__global__ void __launch_bounds__(128)
k_lsh_attn(const float* __restrict__ qk, const float* __restrict__ vv,
           const int* __restrict__ stv, float* __restrict__ so,
           float* __restrict__ slog) {
    __shared__ __align__(16) unsigned short Qs[64 * 64];     // [m][k]
    __shared__ __align__(16) unsigned short Ks[128 * 64];    // [j][k] == Bt for Q.K^T
    __shared__ __align__(16) unsigned short Vts[64 * 128];   // [e][j] == Bt for P.V
    __shared__ __align__(16) unsigned short Ps[64 * 128];    // [m][j]
    __shared__ int tq[64];
    __shared__ int tkv[128];

    int bh = blockIdx.y, ch = blockIdx.x;
    int b = bh / HH, hd = bh % HH;
    int tid = threadIdx.x, wave = tid >> 5, lane = tid & 31;
    int group = lane >> 4, l16 = lane & 15;
    int prev = (ch + NCH - 1) % NCH;

    if (tid < 64) tq[tid] = stv[bh * NHT + ch * 64 + tid];
    else          tkv[tid] = stv[bh * NHT + prev * 64 + (tid - 64)];
    __syncthreads();
    if (tid < 64) tkv[tid] = tq[tid];
    __syncthreads();

    // Q tile, pre-scaled by DH^-1/2
    for (int i = tid; i < 64 * 64; i += 128) {
        int r = i >> 6, e = i & 63;
        float q = qk[((size_t)(b * TT + tq[r])) * DD + hd * DH + e];
        Qs[i] = f2bfu(q * 0.125f);
    }
    // K rows (L2-normalized shared-QK) and V transposed; one row per thread
    {
        int j = tid;                                   // 128 rows, 128 threads
        int tok = tkv[j];
        const float* kr = qk + ((size_t)(b * TT + tok)) * DD + hd * DH;
        const float* vr = vv + ((size_t)(b * TT + tok)) * DD + hd * DH;
        float ss = 0.f;
        for (int e = 0; e < DH; ++e) { float v = kr[e]; ss += v * v; }
        float inv = 1.0f / fmaxf(sqrtf(ss), 1e-12f);
        for (int e = 0; e < DH; ++e) {
            Ks[j * 64 + e]   = f2bfu(kr[e] * inv);
            Vts[e * 128 + j] = f2bfu(vr[e]);
        }
    }
    __syncthreads();

    // ---- S = Q K^T : per wave 16 rows x 128 cols, K-dim 64 ----
    f32x8 sacc[8];
    for (int a = 0; a < 8; ++a)
        for (int i = 0; i < 8; ++i) sacc[a][i] = 0.f;
    int mrow = wave * 16 + l16;

    #pragma unroll
    for (int k0 = 0; k0 < 64; k0 += 32) {
        union { bf16x16 v; u32x4 q[2]; } af;
        af.q[0] = *(const u32x4*)&Qs[mrow * 64 + k0 + group * 8];
        af.q[1] = *(const u32x4*)&Qs[mrow * 64 + k0 + 16 + group * 8];
        #pragma unroll
        for (int a = 0; a < 8; ++a) {
            int n = a * 16 + l16;
            union { bf16x16 v; u32x4 q[2]; } bfv;
            bfv.q[0] = *(const u32x4*)&Ks[n * 64 + k0 + group * 8];
            bfv.q[1] = *(const u32x4*)&Ks[n * 64 + k0 + 16 + group * 8];
            sacc[a] = __builtin_amdgcn_wmma_f32_16x16x32_bf16(
                false, af.v, false, bfv.v, (short)0, sacc[a], false, false);
        }
    }

    // ---- mask + LSE softmax (rows live across 16 lanes of a half-wave) ----
    #pragma unroll
    for (int r = 0; r < 8; ++r) {
        int m = wave * 16 + group * 8 + r;
        int tqm = tq[m];
        float mx = -3.4e38f;
        #pragma unroll
        for (int a = 0; a < 8; ++a) {
            int n = a * 16 + l16;
            float s = sacc[a][r];
            if (tqm == tkv[n]) s = SELF_MASK;
            sacc[a][r] = s;
            mx = fmaxf(mx, s);
        }
        for (int msk = 1; msk < 16; msk <<= 1) mx = fmaxf(mx, __shfl_xor(mx, msk, 32));
        float sum = 0.f;
        #pragma unroll
        for (int a = 0; a < 8; ++a) sum += __expf(sacc[a][r] - mx);
        for (int msk = 1; msk < 16; msk <<= 1) sum += __shfl_xor(sum, msk, 32);
        float lse = mx + __logf(sum);
        #pragma unroll
        for (int a = 0; a < 8; ++a) {
            int n = a * 16 + l16;
            Ps[m * 128 + n] = f2bfu(__expf(sacc[a][r] - lse));
        }
        if (l16 == 0) slog[bh * NHT + ch * 64 + m] = lse;
    }
    __syncthreads();

    // ---- O = P V : per wave 16 rows x 64 cols, K-dim 128 ----
    f32x8 oacc[4];
    for (int a = 0; a < 4; ++a)
        for (int i = 0; i < 8; ++i) oacc[a][i] = 0.f;

    #pragma unroll
    for (int k0 = 0; k0 < 128; k0 += 32) {
        union { bf16x16 v; u32x4 q[2]; } af;
        af.q[0] = *(const u32x4*)&Ps[mrow * 128 + k0 + group * 8];
        af.q[1] = *(const u32x4*)&Ps[mrow * 128 + k0 + 16 + group * 8];
        #pragma unroll
        for (int a = 0; a < 4; ++a) {
            int n = a * 16 + l16;
            union { bf16x16 v; u32x4 q[2]; } bfv;
            bfv.q[0] = *(const u32x4*)&Vts[n * 128 + k0 + group * 8];
            bfv.q[1] = *(const u32x4*)&Vts[n * 128 + k0 + 16 + group * 8];
            oacc[a] = __builtin_amdgcn_wmma_f32_16x16x32_bf16(
                false, af.v, false, bfv.v, (short)0, oacc[a], false, false);
        }
    }
    #pragma unroll
    for (int a = 0; a < 4; ++a) {
        int e = a * 16 + l16;
        #pragma unroll
        for (int r = 0; r < 8; ++r) {
            int m = wave * 16 + group * 8 + r;
            so[((size_t)bh * NHT + ch * 64 + m) * DH + e] = oacc[a][r];
        }
    }
}

// =====================================================================
// 7) Unsort + softmax-combine hash rounds -> bf16 attn in [B,T,D] layout
// =====================================================================
__global__ void k_combine(const float* __restrict__ so, const float* __restrict__ slog,
                          const int* __restrict__ undo, unsigned short* __restrict__ out) {
    int idx = blockIdx.x * 256 + threadIdx.x;       // bh*T*DH
    if (idx >= BHD * TT * DH) return;
    int e  = idx % DH;
    int t  = (idx / DH) % TT;
    int bh = idx / (DH * TT);
    int b = bh / HH, hd = bh % HH;
    float lg[NHASH]; int pos[NHASH];
    float mx = -3.4e38f;
    #pragma unroll
    for (int nh = 0; nh < NHASH; ++nh) {
        int p = undo[bh * NHT + nh * TT + t];
        pos[nh] = p;
        float l = slog[bh * NHT + p];
        lg[nh] = l;
        mx = fmaxf(mx, l);
    }
    float sum = 0.f;
    #pragma unroll
    for (int nh = 0; nh < NHASH; ++nh) sum += __expf(lg[nh] - mx);
    float o = 0.f;
    #pragma unroll
    for (int nh = 0; nh < NHASH; ++nh)
        o += (__expf(lg[nh] - mx) / sum) * so[((size_t)bh * NHT + pos[nh]) * DH + e];
    out[((size_t)(b * TT + t)) * DD + hd * DH + e] = f2bfu(o);
}

// =====================================================================
// 8) enc = 0.5*(x1+x2); prefix sums over T  (thread per (b,d), coalesced)
// =====================================================================
__global__ void k_cumsum(const float* __restrict__ x1, const float* __restrict__ x2,
                         float* __restrict__ cs) {
    int idx = blockIdx.x * 256 + threadIdx.x;       // B*D
    if (idx >= BB * DD) return;
    int d = idx % DD, b = idx / DD;
    float run = 0.f;
    cs[((size_t)b * (TT + 1)) * DD + d] = 0.f;
    for (int t = 0; t < TT; ++t) {
        size_t p = ((size_t)b * TT + t) * DD + d;
        run += 0.5f * (x1[p] + x2[p]);
        cs[((size_t)b * (TT + 1) + t + 1) * DD + d] = run;
    }
}

// =====================================================================
// 9) Ragged span mean pooling via prefix sums
// =====================================================================
__global__ void k_spans(const float* __restrict__ cs, const int* __restrict__ starts,
                        const int* __restrict__ lens, float* __restrict__ out) {
    int idx = blockIdx.x * 256 + threadIdx.x;       // B*S*D
    if (idx >= BB * SS * DD) return;
    int d = idx % DD;
    int s = (idx / DD) % SS;
    int b = idx / (DD * SS);
    int st0 = starts[b * SS + s];
    int len = lens[b * SS + s] + 1;
    int en  = st0 + len;
    float v = cs[((size_t)b * (TT + 1) + en) * DD + d]
            - cs[((size_t)b * (TT + 1) + st0) * DD + d];
    out[idx] = v / (float)len;
}

// =====================================================================
// host orchestration
// =====================================================================
extern "C" void kernel_launch(void* const* d_in, const int* in_sizes, int n_in,
                              void* d_out, int out_size, void* d_ws, size_t ws_size,
                              hipStream_t stream) {
    const float* x       = (const float*)d_in[0];
    const float* conv_w  = (const float*)d_in[1];
    const float* conv_b  = (const float*)d_in[2];
    const float* rots    = (const float*)d_in[3];
    const float* ln1_g   = (const float*)d_in[4];
    const float* ln1_b   = (const float*)d_in[5];
    const float* wqk     = (const float*)d_in[6];
    const float* wv      = (const float*)d_in[7];
    const float* wo      = (const float*)d_in[8];
    const float* bo      = (const float*)d_in[9];
    const float* ln2_g   = (const float*)d_in[10];
    const float* ln2_b   = (const float*)d_in[11];
    const float* w1      = (const float*)d_in[12];
    const float* b1      = (const float*)d_in[13];
    const float* w2      = (const float*)d_in[14];
    const float* b2      = (const float*)d_in[15];
    const int*   sp_st   = (const int*)d_in[16];
    const int*   sp_len  = (const int*)d_in[17];
    float* out = (float*)d_out;

    // ---- workspace carve-up ----
    size_t off = 0;
    char* base = (char*)d_ws;
    auto alloc = [&](size_t bytes) {
        void* p = base + off;
        off += (bytes + 255) & ~(size_t)255;
        return p;
    };
    const size_t act = (size_t)MROWS * DD * sizeof(float);                  // 16 MB
    float* x1    = (float*)alloc(act);
    float* x2    = (float*)alloc(act);
    float* qkf   = (float*)alloc(act);
    float* vvf   = (float*)alloc(act);
    float* cs    = (float*)alloc((size_t)BB * (TT + 1) * DD * sizeof(float));
    unsigned short* lnb16  = (unsigned short*)alloc((size_t)MROWS * DD * 2);
    unsigned short* attn16 = (unsigned short*)alloc((size_t)MROWS * DD * 2);
    unsigned short* wqk16  = (unsigned short*)alloc((size_t)DD * DD * 2);
    unsigned short* wv16   = (unsigned short*)alloc((size_t)DD * DD * 2);
    unsigned short* wo16   = (unsigned short*)alloc((size_t)DD * DD * 2);
    unsigned short* w116   = (unsigned short*)alloc((size_t)DD * DFF * 2);
    unsigned short* w216   = (unsigned short*)alloc((size_t)DFF * DD * 2);
    int*   bkts  = (int*)alloc((size_t)BHD * NHT * sizeof(int));
    int*   stv   = (int*)alloc((size_t)BHD * NHT * sizeof(int));
    int*   undo  = (int*)alloc((size_t)BHD * NHT * sizeof(int));
    int*   Hc    = (int*)alloc((size_t)BHD * 64 * 256 * sizeof(int));
    int*   TOFF  = (int*)alloc((size_t)BHD * 64 * 256 * sizeof(int));
    float* slog  = (float*)alloc((size_t)BHD * NHT * sizeof(float));
    // union: sorted attention out (bh*NHT*DH f32, 64MB) and FFN bf16 intermediate (32MB)
    float* big   = (float*)alloc((size_t)BHD * NHT * DH * sizeof(float));
    float* so    = big;
    unsigned short* ff16 = (unsigned short*)big;
    if (off > ws_size) return;   // deterministic no-op if scratch too small

    // ---- one-time weight convert + transpose to bf16 [N][K] ----
    k_wT_bf16<<<(DD * DD + 255) / 256, 256, 0, stream>>>(wqk, wqk16, DD, DD);
    k_wT_bf16<<<(DD * DD + 255) / 256, 256, 0, stream>>>(wv,  wv16,  DD, DD);
    k_wT_bf16<<<(DD * DD + 255) / 256, 256, 0, stream>>>(wo,  wo16,  DD, DD);
    k_wT_bf16<<<(DD * DFF + 255) / 256, 256, 0, stream>>>(w1, w116, DD, DFF);
    k_wT_bf16<<<(DFF * DD + 255) / 256, 256, 0, stream>>>(w2, w216, DFF, DD);

    // ---- embedding ----
    k_embed<<<(BB * TT * DD + 255) / 256, 256, 0, stream>>>(x, conv_w, conv_b, x1, x2);

    dim3 gLN((MROWS + 7) / 8);
    dim3 gP512(DD / GBN, MROWS / GBM);     // (8, 128)
    dim3 gFF1(DFF / GBN, MROWS / GBM);     // (32, 128)
    dim3 gHash((BHD * NHASH * TT + 255) / 256);
    dim3 gSort(64, BHD);
    dim3 gAttn(NCH, BHD);
    dim3 gComb((BHD * TT * DH + 255) / 256);

    for (int dep = 0; dep < DEPTH; ++dep) {
        // x1 += lsh_attention(LN(x2))
        k_layernorm<<<gLN, 256, 0, stream>>>(x2, ln1_g, ln1_b, lnb16);
        k_gemm_async<0, 0><<<gP512, 128, 0, stream>>>(lnb16, wqk16, nullptr, nullptr, qkf, MROWS, DD, DD);
        k_gemm_async<0, 0><<<gP512, 128, 0, stream>>>(lnb16, wv16,  nullptr, nullptr, vvf, MROWS, DD, DD);
        k_hash<<<gHash, 256, 0, stream>>>(qkf, rots, bkts);
        k_hist<<<gSort, 256, 0, stream>>>(bkts, Hc);
        k_binoff<<<BHD, 256, 0, stream>>>(Hc, TOFF);
        k_scatter<<<gSort, 256, 0, stream>>>(bkts, TOFF, stv, undo);
        k_lsh_attn<<<gAttn, 128, 0, stream>>>(qkf, vvf, stv, so, slog);
        k_combine<<<gComb, 256, 0, stream>>>(so, slog, undo, attn16);
        k_gemm_async<3, 0><<<gP512, 128, 0, stream>>>(attn16, wo16, bo, x1, x1, MROWS, DD, DD);
        // x2 += FFN(LN(x1))
        k_layernorm<<<gLN, 256, 0, stream>>>(x1, ln2_g, ln2_b, lnb16);
        k_gemm_async<2, 1><<<gFF1, 128, 0, stream>>>(lnb16, w116, b1, nullptr, ff16, MROWS, DFF, DD);
        k_gemm_async<3, 0><<<gP512, 128, 0, stream>>>(ff16, w216, b2, x2, x2, MROWS, DD, DFF);
    }

    k_cumsum<<<(BB * DD + 255) / 256, 256, 0, stream>>>(x1, x2, cs);
    k_spans<<<(BB * SS * DD + 255) / 256, 256, 0, stream>>>(cs, sp_st, sp_len, out);
    (void)n_in; (void)in_sizes; (void)out_size;
}
